// BiMPNNLayer_2662879724349
// MI455X (gfx1250) — matrix operand order repaired and checked by reading, not compiled, not run on hardware
//
#include <hip/hip_runtime.h>
#include <cmath>

typedef __attribute__((ext_vector_type(16))) _Float16 v16h;
typedef __attribute__((ext_vector_type(8)))  _Float16 v8h;
typedef __attribute__((ext_vector_type(4)))  _Float16 v4h;
typedef __attribute__((ext_vector_type(8)))  float    v8f;

#define DIMF 128
#define LDSW 136   // 128 + 8 half padding -> rotates LDS banks per row

// ---------------------------------------------------------------------------
// Kernel 1: fused triple GEMM.  Wh = h W^T + b (f16), Wth = h Wt^T + bt (f16),
// Wsh = h Ws^T + bs (f32, written straight into d_out as the accumulator base).
// One wave computes a 16-row x 128-col tile via 3*8*4 = 96 v_wmma ops.
// ---------------------------------------------------------------------------
__global__ __launch_bounds__(256) void bimpnn_gemm3(
    const float* __restrict__ h,
    const float* __restrict__ Ww,  const float* __restrict__ Wb,
    const float* __restrict__ Wtw, const float* __restrict__ Wtb,
    const float* __restrict__ Wsw, const float* __restrict__ Wsb,
    _Float16* __restrict__ Wh, _Float16* __restrict__ Wth,
    float* __restrict__ Wsh, int n)
{
    __shared__ _Float16 lw[3 * 128 * LDSW];   // 104448 B of the 320KB WGP LDS

    // Stage weights fp32 -> f16 into padded LDS (once per block)
    for (int idx = threadIdx.x; idx < 3 * 128 * 128; idx += 256) {
        int m   = idx >> 14;
        int rem = idx & 16383;
        int o   = rem >> 7;
        int k   = rem & 127;
        const float* src = (m == 0) ? Ww : ((m == 1) ? Wtw : Wsw);
        lw[(m * 128 + o) * LDSW + k] = (_Float16)src[rem];
    }
    __syncthreads();

    const int wave = threadIdx.x >> 5;
    const int lane = threadIdx.x & 31;
    const int tile = (blockIdx.x * 8 + wave) * 16;      // first row of tile
    if (tile >= n) return;                               // no barriers follow

    const int colLane = lane & 15;

    // Preload all bias values (3 mats x 8 column tiles) -> no loads in hot loop
    float bias0[8], bias1[8], bias2[8];
#pragma unroll
    for (int nt = 0; nt < 8; ++nt) {
        bias0[nt] = Wb [nt * 16 + colLane];
        bias1[nt] = Wtb[nt * 16 + colLane];
        bias2[nt] = Wsb[nt * 16 + colLane];
    }

    // ---- A fragments: rows tile..tile+15 of h, converted to f16 -----------
    // ISA layout (16-bit A 16x32): lane&15 = M; lanes<16 hold K {0..7,16..23},
    // lanes>=16 hold K {8..15,24..31}.
    int mrow = tile + colLane;
    if (mrow > n - 1) mrow = n - 1;                      // clamp, store-guarded
    const int ksel = (lane >> 4) * 8;

    v16h afrag[4];
#pragma unroll
    for (int kk = 0; kk < 4; ++kk) {
        const float* p = h + (size_t)mrow * DIMF + kk * 32 + ksel;
        float4 c0 = *(const float4*)(p);
        float4 c1 = *(const float4*)(p + 4);
        float4 c2 = *(const float4*)(p + 16);
        float4 c3 = *(const float4*)(p + 20);
        v16h a;
        a[0]=(_Float16)c0.x;  a[1]=(_Float16)c0.y;  a[2]=(_Float16)c0.z;  a[3]=(_Float16)c0.w;
        a[4]=(_Float16)c1.x;  a[5]=(_Float16)c1.y;  a[6]=(_Float16)c1.z;  a[7]=(_Float16)c1.w;
        a[8]=(_Float16)c2.x;  a[9]=(_Float16)c2.y;  a[10]=(_Float16)c2.z; a[11]=(_Float16)c2.w;
        a[12]=(_Float16)c3.x; a[13]=(_Float16)c3.y; a[14]=(_Float16)c3.z; a[15]=(_Float16)c3.w;
        afrag[kk] = a;
    }

    const int rbase = tile + ((lane >> 4) * 8);          // C rows: M = r + 8*(lane>=16)
    const bool full = (tile + 16 <= n);                  // whole tile in range?

#pragma unroll
    for (int m = 0; m < 3; ++m) {
        const _Float16* wbase = &lw[m * 128 * LDSW];
        for (int nt = 0; nt < 8; ++nt) {
            const int col = nt * 16 + colLane;
            const float bv = (m == 0) ? bias0[nt] : ((m == 1) ? bias1[nt] : bias2[nt]);
            v8f acc = {bv, bv, bv, bv, bv, bv, bv, bv};

            // B fragment (32x16 f16): lane = column, lanes<16 hold K {0..7,16..23},
            // lanes>=16 hold K {8..15,24..31} -> two 8-half runs of W row `col`.
            const _Float16* bp = wbase + col * LDSW + ((lane >> 4) * 8);
#pragma unroll
            for (int kk = 0; kk < 4; ++kk) {
                v8h lo = *(const v8h*)(bp + kk * 32);
                v8h hi = *(const v8h*)(bp + kk * 32 + 16);
                v16h b;
#pragma unroll
                for (int i = 0; i < 8; ++i) { b[i] = lo[i]; b[i + 8] = hi[i]; }
                acc = __builtin_amdgcn_wmma_f32_16x16x32_f16(
                          false, afrag[kk], false, b, (short)0, acc, false, false);
            }

            if (full) {
                // Fast path: straight-line stores, no exec-mask juggling
                if (m == 0) {
#pragma unroll
                    for (int r = 0; r < 8; ++r)
                        Wh [(size_t)(rbase + r) * DIMF + col] = (_Float16)acc[r];
                } else if (m == 1) {
#pragma unroll
                    for (int r = 0; r < 8; ++r)
                        Wth[(size_t)(rbase + r) * DIMF + col] = (_Float16)acc[r];
                } else {
#pragma unroll
                    for (int r = 0; r < 8; ++r)
                        Wsh[(size_t)(rbase + r) * DIMF + col] = acc[r];
                }
            } else {
                // Tail tile only: guarded stores
                for (int r = 0; r < 8; ++r) {
                    const int row = rbase + r;
                    if (row < n) {
                        if (m == 0)      Wh [(size_t)row * DIMF + col] = (_Float16)acc[r];
                        else if (m == 1) Wth[(size_t)row * DIMF + col] = (_Float16)acc[r];
                        else             Wsh[(size_t)row * DIMF + col] = acc[r];
                    }
                }
            }
        }
    }
}

// ---------------------------------------------------------------------------
// CSR construction: zero -> histogram -> scan (2 blocks) -> fill
// ---------------------------------------------------------------------------
__global__ void bimpnn_zero_i32(int* __restrict__ p, int cnt)
{
    int i = blockIdx.x * blockDim.x + threadIdx.x;
    if (i < cnt) p[i] = 0;
}

__global__ void bimpnn_hist(const int* __restrict__ rows, const int* __restrict__ cols,
                            int* __restrict__ deg_r, int* __restrict__ deg_c, int E)
{
    int e = blockIdx.x * blockDim.x + threadIdx.x;
    if (e < E) {
        atomicAdd(&deg_r[rows[e]], 1);
        atomicAdd(&deg_c[cols[e]], 1);
    }
}

__global__ __launch_bounds__(1024) void bimpnn_scan(
    const int* __restrict__ deg_r, int* __restrict__ off_r, int* __restrict__ cur_r,
    const int* __restrict__ deg_c, int* __restrict__ off_c, int* __restrict__ cur_c,
    int n)
{
    __shared__ int lds[1024];
    const int* deg = blockIdx.x ? deg_c : deg_r;
    int* off       = blockIdx.x ? off_c : off_r;
    int* cur       = blockIdx.x ? cur_c : cur_r;

    const int t = threadIdx.x;
    const int chunk = (n + 1023) >> 10;
    const int s = t * chunk;
    const int e = min(s + chunk, n);

    int sum = 0;
    for (int i = s; i < e; ++i) sum += deg[i];
    lds[t] = sum;
    __syncthreads();
    for (int o = 1; o < 1024; o <<= 1) {
        int v = lds[t];
        if (t >= o) v += lds[t - o];
        __syncthreads();
        lds[t] = v;
        __syncthreads();
    }
    int run = (t == 0) ? 0 : lds[t - 1];
    for (int i = s; i < e; ++i) {
        off[i] = run; cur[i] = run;
        run += deg[i];
    }
    if (e == n && s < n) off[n] = run;
}

__global__ void bimpnn_fill(const int* __restrict__ rows, const int* __restrict__ cols,
                            int* __restrict__ cur_r, int* __restrict__ cur_c,
                            int* __restrict__ csr_r, int* __restrict__ csr_c, int E)
{
    int e = blockIdx.x * blockDim.x + threadIdx.x;
    if (e < E) {
        int r = rows[e], c = cols[e];
        int p1 = atomicAdd(&cur_r[r], 1); csr_r[p1] = c;   // neighbors of row r
        int p2 = atomicAdd(&cur_c[c], 1); csr_c[p2] = r;   // neighbors of col c
    }
}

// ---------------------------------------------------------------------------
// Gather + exact GELU. One wave per node; lane owns 4 features (8B f16 reads
// -> 256B coalesced per neighbor row, L2-resident since Wh/Wth are f16).
// Unrolled x2 to keep two gather loads in flight. d_out already holds Wsh.
// ---------------------------------------------------------------------------
__global__ __launch_bounds__(256) void bimpnn_gather(
    const _Float16* __restrict__ Wh, const _Float16* __restrict__ Wth,
    const int* __restrict__ off_r, const int* __restrict__ csr_r,
    const int* __restrict__ off_c, const int* __restrict__ csr_c,
    float* __restrict__ out, int n)
{
    int gid  = blockIdx.x * blockDim.x + threadIdx.x;
    int node = gid >> 5;
    int lane = gid & 31;
    if (node >= n) return;

    const int fo = lane * 4;
    float* op = out + (size_t)node * DIMF + fo;
    float4 acc = *(const float4*)op;                      // Wsh + biases

    int s = off_r[node], e = off_r[node + 1];
    int j = s;
    for (; j + 1 < e; j += 2) {
        int nb0 = csr_r[j], nb1 = csr_r[j + 1];
        v4h v0 = *(const v4h*)(Wh + (size_t)nb0 * DIMF + fo);
        v4h v1 = *(const v4h*)(Wh + (size_t)nb1 * DIMF + fo);
        acc.x += (float)v0[0] + (float)v1[0];
        acc.y += (float)v0[1] + (float)v1[1];
        acc.z += (float)v0[2] + (float)v1[2];
        acc.w += (float)v0[3] + (float)v1[3];
    }
    if (j < e) {
        int nb = csr_r[j];
        v4h v = *(const v4h*)(Wh + (size_t)nb * DIMF + fo);
        acc.x += (float)v[0]; acc.y += (float)v[1];
        acc.z += (float)v[2]; acc.w += (float)v[3];
    }

    s = off_c[node]; e = off_c[node + 1];
    j = s;
    for (; j + 1 < e; j += 2) {
        int nb0 = csr_c[j], nb1 = csr_c[j + 1];
        v4h v0 = *(const v4h*)(Wth + (size_t)nb0 * DIMF + fo);
        v4h v1 = *(const v4h*)(Wth + (size_t)nb1 * DIMF + fo);
        acc.x += (float)v0[0] + (float)v1[0];
        acc.y += (float)v0[1] + (float)v1[1];
        acc.z += (float)v0[2] + (float)v1[2];
        acc.w += (float)v0[3] + (float)v1[3];
    }
    if (j < e) {
        int nb = csr_c[j];
        v4h v = *(const v4h*)(Wth + (size_t)nb * DIMF + fo);
        acc.x += (float)v[0]; acc.y += (float)v[1];
        acc.z += (float)v[2]; acc.w += (float)v[3];
    }

    const float k = 0.70710678118654752f;                 // 1/sqrt(2)
    acc.x = 0.5f * acc.x * (1.0f + erff(acc.x * k));
    acc.y = 0.5f * acc.y * (1.0f + erff(acc.y * k));
    acc.z = 0.5f * acc.z * (1.0f + erff(acc.z * k));
    acc.w = 0.5f * acc.w * (1.0f + erff(acc.w * k));
    *(float4*)op = acc;
}

// ---------------------------------------------------------------------------
extern "C" void kernel_launch(void* const* d_in, const int* in_sizes, int n_in,
                              void* d_out, int out_size, void* d_ws, size_t ws_size,
                              hipStream_t stream)
{
    const float* h_n  = (const float*)d_in[0];
    const float* Ww   = (const float*)d_in[1];
    const float* Wb   = (const float*)d_in[2];
    const float* Wtw  = (const float*)d_in[3];
    const float* Wtb  = (const float*)d_in[4];
    const float* Wsw  = (const float*)d_in[5];
    const float* Wsb  = (const float*)d_in[6];
    const int*   rows = (const int*)d_in[7];
    const int*   cols = (const int*)d_in[8];
    float* out = (float*)d_out;

    const int N = in_sizes[0] / DIMF;
    const int E = in_sizes[7];

    // Workspace carve-up (256B aligned)
    char* base = (char*)d_ws;
    size_t off = 0;
    auto carve = [&](size_t bytes) {
        char* p = base + off;
        off = (off + bytes + 255) & ~(size_t)255;
        return p;
    };
    _Float16* Wh    = (_Float16*)carve((size_t)N * DIMF * 2);
    _Float16* Wth   = (_Float16*)carve((size_t)N * DIMF * 2);
    int* deg        = (int*)carve((size_t)2 * N * 4);      // deg_r | deg_c contiguous
    int* deg_r = deg;
    int* deg_c = deg + N;
    int* off_r      = (int*)carve((size_t)(N + 1) * 4);
    int* off_c      = (int*)carve((size_t)(N + 1) * 4);
    int* cur_r      = (int*)carve((size_t)N * 4);
    int* cur_c      = (int*)carve((size_t)N * 4);
    int* csr_r      = (int*)carve((size_t)E * 4);
    int* csr_c      = (int*)carve((size_t)E * 4);
    (void)ws_size;

    // 1) GEMMs: Wh/Wth -> f16 ws, Wsh -> d_out (f32 accumulator base)
    bimpnn_gemm3<<<(N + 127) / 128, 256, 0, stream>>>(
        h_n, Ww, Wb, Wtw, Wtb, Wsw, Wsb, Wh, Wth, out, N);

    // 2) CSR build
    bimpnn_zero_i32<<<(2 * N + 255) / 256, 256, 0, stream>>>(deg, 2 * N);
    bimpnn_hist<<<(E + 255) / 256, 256, 0, stream>>>(rows, cols, deg_r, deg_c, E);
    bimpnn_scan<<<2, 1024, 0, stream>>>(deg_r, off_r, cur_r, deg_c, off_c, cur_c, N);
    bimpnn_fill<<<(E + 255) / 256, 256, 0, stream>>>(rows, cols, cur_r, cur_c, csr_r, csr_c, E);

    // 3) Atomic-free gather + exact GELU (in place on d_out)
    bimpnn_gather<<<(N + 7) / 8, 256, 0, stream>>>(
        Wh, Wth, off_r, csr_r, off_c, csr_c, out, N);
}
